// GNN_58394375356578
// MI455X (gfx1250) — matrix operand order repaired and checked by reading, not compiled, not run on hardware
//
#include <hip/hip_runtime.h>

// ---------------------------------------------------------------------------
// MI455X (gfx1250) chemprop-style MPNN forward.
//  - All GEMMs via v_wmma_f32_16x16x32_bf16 (wave32), fp32 accumulate.
//  - Weights pre-packed to bf16 pairs once; B tiles of the dominant GEMM are
//    moved by the Tensor Data Mover (tensor_load_to_lds, double-buffered,
//    TENSORcnt-synchronized) with D# LDS padding matching the [64][17] layout.
//  - A tiles (computed operands) staged via VALU fp32->bf16 packing.
// Roofline: ~23 GFLOP total vs ~500 MB HBM traffic (~21 us @ 23.3 TB/s);
// bf16 WMMA keeps matrix work at the same order as memory time, while fp32
// WMMA (16x16x4) would be ~8x more matrix instructions and compute-bound.
// ---------------------------------------------------------------------------

typedef __attribute__((ext_vector_type(16))) __bf16 v16bf;
typedef __attribute__((ext_vector_type(8)))  float  v8f;
typedef unsigned int u32x4 __attribute__((ext_vector_type(4)));
typedef int          i32x4 __attribute__((ext_vector_type(4)));
typedef int          i32x8 __attribute__((ext_vector_type(8)));

union Frag { v16bf v; unsigned int u[8]; };

#define BSd 16          // B*S
#define Nd  64          // atoms
#define Hd  256         // hidden
#define AFd 133
#define BFd 147
#define MB_BS_STRIDE (Nd * Nd * Hd)      // 1048576
#define MB_N_STRIDE (Nd * Hd)            // 16384
// packed bf16 pair counts per weight row
#define PA 67           // ceil(133/2)
#define PB 74           // ceil(147/2)
#define PH 128          // 256/2
#define PO 384          // 768/2

// round-to-nearest-even float -> bf16, packed pair (no __bf16 scalar math)
__device__ __forceinline__ unsigned int pack2bf(float a, float b) {
  union { float f; unsigned int u; } x, y;
  x.f = a; y.f = b;
  unsigned int ra = (x.u + 0x7FFFu + ((x.u >> 16) & 1u)) >> 16;
  unsigned int rb = (y.u + 0x7FFFu + ((y.u >> 16) & 1u)) >> 16;
  return (ra & 0xFFFFu) | (rb << 16);
}

// low 32 bits of a flat pointer to __shared__ == wave-relative LDS byte offset
__device__ __forceinline__ unsigned int lds_off(const void* p) {
  return (unsigned int)(unsigned long long)p;
}

// TDM: DMA a 64-row x 16-DWORD tile (row stride 128 DWORDs) from global bf16
// weights into LDS, inserting 1 pad DWORD after every 16 DWORDs (-> [64][17]).
__device__ __forceinline__ void tdm_load_w(unsigned int lds_addr, const unsigned int* gsrc) {
  unsigned long long ga = (unsigned long long)gsrc;
  u32x4 g0;
  g0[0] = 1u;                                                // count=1, load
  g0[1] = lds_addr;                                          // LDS dest (bytes)
  g0[2] = (unsigned int)(ga & 0xFFFFFFFFu);                  // global_addr lo
  g0[3] = (unsigned int)((ga >> 32) & 0x01FFFFFFu) | 0x80000000u; // hi | type=2
  i32x8 g1;
  g1[0] = (int)((2u << 16) | (1u << 20) | (3u << 22));       // 4B elem, pad_en,
                                                             // interval=16DW, pad=1DW
  g1[1] = (int)(16u << 16);                                  // tensor_dim0 = 16
  g1[2] = (int)(64u << 16);                                  // tensor_dim1 = 64
  g1[3] = (int)(16u << 16);                                  // tile_dim0   = 16
  g1[4] = 64;                                                // tile_dim1   = 64
  g1[5] = 128;                                               // dim0_stride = 128 DW
  g1[6] = 0; g1[7] = 0;
  i32x4 gz = {0, 0, 0, 0};
#if defined(__clang_major__) && (__clang_major__ >= 23)
  i32x8 gz8 = {0, 0, 0, 0, 0, 0, 0, 0};
  __builtin_amdgcn_tensor_load_to_lds(g0, g1, gz, gz, gz8, 0);
#else
  __builtin_amdgcn_tensor_load_to_lds(g0, g1, gz, gz, 0);
#endif
}

// One K=32 slab: A strip is 16 rows (wrow..wrow+15) x 32 K, B is 64 cols x 32 K.
// lA/lB hold packed bf16 pairs (uint = 2 K-values), row strides in uints,
// kOff in uints (pairs). Produces 4 WMMAs covering 16(M) x 64(N).
__device__ __forceinline__ void wmma_4x(const unsigned int* lA, int sA,
                                        const unsigned int* lB, int sB,
                                        int kOff, int lane, int wrow, v8f* acc) {
  const int lr = lane & 15, kh = lane >> 4;
  Frag a;
  const unsigned int* ap = lA + (wrow + lr) * sA + kOff;
#pragma unroll
  for (int j = 0; j < 8; ++j)
    a.u[j] = ap[((j & 4) << 1) + (kh << 2) + (j & 3)];   // A 16-bit lane layout
#pragma unroll
  for (int c = 0; c < 4; ++c) {
    Frag b;
    const unsigned int* bp = lB + ((c << 4) + lr) * sB + kOff + (kh << 2);
#pragma unroll
    for (int j = 0; j < 8; ++j) b.u[j] = bp[((j & 4) << 1) + (j & 3)];  // B 16-bit lane layout
    acc[c] = __builtin_amdgcn_wmma_f32_16x16x32_bf16(
        false, a.v, false, b.v, (short)0, acc[c], false, false);
  }
}

// Stage a 64-row x 32-K fp32 tile (zero-pad K >= kmax) as packed bf16 pairs.
__device__ __forceinline__ void stage64x32(unsigned int (*dst)[17],
                                           const float* __restrict__ src,
                                           int stride, int k0, int kmax, int tid) {
  for (int p = tid; p < 1024; p += 128) {
    int r = p >> 4, kp = p & 15;
    int k = k0 + (kp << 1);
    float a = (k     < kmax) ? src[r * stride + k]     : 0.f;
    float b = (k + 1 < kmax) ? src[r * stride + k + 1] : 0.f;
    dst[r][kp] = pack2bf(a, b);
  }
}

// Stage a 64-row x 16-pair tile from pre-packed bf16 weights (pure copy).
__device__ __forceinline__ void stage64x32_bf(unsigned int (*dst)[17],
                                              const unsigned int* __restrict__ src,
                                              int stridePairs, int kp0, int maxPairs,
                                              int tid) {
  for (int p = tid; p < 1024; p += 128) {
    int r = p >> 4, kp = p & 15, k = kp0 + kp;
    dst[r][kp] = (k < maxPairs) ? src[r * stridePairs + k] : 0u;
  }
}

// --------- pack a fp32 [rows,K] weight into bf16 pairs [rows,pairs] ---------
__global__ void __launch_bounds__(256)
k_cvt(const float* __restrict__ W, unsigned int* __restrict__ out,
      int K, int pairs, int total) {
  int i = blockIdx.x * 256 + threadIdx.x;
  if (i >= total) return;
  int r = i / pairs, kp = i - r * pairs, k = kp << 1;
  float a = (k     < K) ? W[r * K + k]     : 0.f;
  float b = (k + 1 < K) ? W[r * K + k + 1] : 0.f;
  out[i] = pack2bf(a, b);
}

// ------------------- input_atom = relu(f_atoms @ W_i_atom^T) ----------------
__global__ void __launch_bounds__(128)
k_in_atom(const float* __restrict__ fa, const unsigned int* __restrict__ WiBf,
          float* __restrict__ inAtom, float* __restrict__ msgA) {
  __shared__ unsigned int lA[64][17], lB[64][17];
  const int tid = threadIdx.x, lane = tid & 31, wave = tid >> 5;
  const int m0 = blockIdx.x * 64, n0 = blockIdx.y * 64;
  v8f acc[4] = {};
  for (int kc = 0; kc < 5; ++kc) {              // ceil(133/32)
    stage64x32(lA, fa + (long)m0 * AFd, AFd, kc * 32, AFd, tid);
    stage64x32_bf(lB, WiBf + (long)n0 * PA, PA, kc * 16, PA, tid);
    __syncthreads();
    wmma_4x(&lA[0][0], 17, &lB[0][0], 17, 0, lane, wave * 16, acc);
    __syncthreads();
  }
  const int lr = lane & 15, hi = lane >> 4;
#pragma unroll
  for (int c = 0; c < 4; ++c)
#pragma unroll
    for (int v = 0; v < 8; ++v) {
      int row = m0 + wave * 16 + v + (hi << 3);
      int g   = n0 + (c << 4) + lr;
      float val = acc[c][v]; val = val > 0.f ? val : 0.f;
      inAtom[row * Hd + g] = val;
      msgA  [row * Hd + g] = val;
    }
}

// -------- input_bond = adj * relu(f_bonds @ W_i_bond^T)  (adj binary) -------
__global__ void __launch_bounds__(128)
k_in_bond(const float* __restrict__ fb, const unsigned int* __restrict__ WiBf,
          const float* __restrict__ adj, float* __restrict__ inBond) {
  __shared__ unsigned int lA[64][17], lB[64][17];
  const int tid = threadIdx.x, lane = tid & 31, wave = tid >> 5;
  const int m0 = blockIdx.x * 64, n0 = blockIdx.y * 64;
  v8f acc[4] = {};
  for (int kc = 0; kc < 5; ++kc) {              // ceil(147/32)
    stage64x32(lA, fb + (long)m0 * BFd, BFd, kc * 32, BFd, tid);
    stage64x32_bf(lB, WiBf + (long)n0 * PB, PB, kc * 16, PB, tid);
    __syncthreads();
    wmma_4x(&lA[0][0], 17, &lB[0][0], 17, 0, lane, wave * 16, acc);
    __syncthreads();
  }
  const int lr = lane & 15, hi = lane >> 4;
#pragma unroll
  for (int c = 0; c < 4; ++c)
#pragma unroll
    for (int v = 0; v < 8; ++v) {
      int row = m0 + wave * 16 + v + (hi << 3);
      int g   = n0 + (c << 4) + lr;
      float val = acc[c][v]; val = val > 0.f ? val : 0.f;
      inBond[(long)row * Hd + g] = val * adj[row];
    }
}

// --- agg[bs,m,h] = sum_n mb * sigmoid(max_n mb); mode 0: msgA+=, 1: out -----
__global__ void __launch_bounds__(256)
k_agg(const float* __restrict__ mb, float* __restrict__ msgA,
      float* __restrict__ aggOut, int mode) {
  const int bsm = blockIdx.x;                 // bs*64 + m
  const int bs = bsm >> 6, m = bsm & 63, h = threadIdx.x;
  const float* p = mb + (long)bs * MB_BS_STRIDE + m * Hd + h;
  float s = 0.f, mx = -3.4e38f;
#pragma unroll 4
  for (int n = 0; n < Nd; ++n) {
    float v = p[(long)n * MB_N_STRIDE];
    s += v; mx = fmaxf(mx, v);
  }
  float a = s * (1.f / (1.f + __expf(-mx)));
  const int idx = bsm * Hd + h;
  if (mode == 0) msgA[idx] += a; else aggOut[idx] = a;
}

// ---- soft[bs,n,m] = softmax_n( (msgA[n]·msgA[m]) * adj[bs,n,m] ) -----------
__global__ void __launch_bounds__(128)
k_resonance(const float* __restrict__ msgA, const float* __restrict__ adj,
            float* __restrict__ soft) {
  __shared__ unsigned int lM[64][129];        // 64 x 256 bf16 (packed pairs)
  __shared__ float resL[64][65];
  const int bs = blockIdx.x;
  const int tid = threadIdx.x, lane = tid & 31, wave = tid >> 5;
  const float* src = msgA + (long)bs * Nd * Hd;
  for (int p = tid; p < 8192; p += 128) {
    int r = p >> 7, kp = p & 127;
    lM[r][kp] = pack2bf(src[r * Hd + 2 * kp], src[r * Hd + 2 * kp + 1]);
  }
  __syncthreads();
  v8f acc[4] = {};
#pragma unroll
  for (int kc = 0; kc < 8; ++kc)              // K = 256 = 8 x 32
    wmma_4x(&lM[0][0], 129, &lM[0][0], 129, kc * 16, lane, wave * 16, acc);
  const int lr = lane & 15, hi = lane >> 4;
  const float* adjp = adj + bs * Nd * Nd;
#pragma unroll
  for (int c = 0; c < 4; ++c)
#pragma unroll
    for (int v = 0; v < 8; ++v) {
      int n = wave * 16 + v + (hi << 3);
      int m = (c << 4) + lr;
      resL[n][m] = acc[c][v] * adjp[n * Nd + m];
    }
  __syncthreads();
  if (tid < 64) {                             // softmax over n for column m
    const int m = tid;
    float mx = -3.4e38f;
    for (int n = 0; n < Nd; ++n) mx = fmaxf(mx, resL[n][m]);
    float s = 0.f;
    for (int n = 0; n < Nd; ++n) { float e = __expf(resL[n][m] - mx); resL[n][m] = e; s += e; }
    const float inv = 1.f / s;
    for (int n = 0; n < Nd; ++n) soft[bs * Nd * Nd + n * Nd + m] = resL[n][m] * inv;
  }
}

// ---- message step: mbNext[bs,n,m,:] =
//   relu(inBond + (adj[m,n]*msgA[n,:] - mbCur[bs,m,n,:]) @ Wh^T) * soft[n,m]
// B tiles DMA'd by the TDM, double-buffered on TENSORcnt.
__global__ void __launch_bounds__(128)
k_step(const float* __restrict__ mbCur, const float* __restrict__ msgA,
       const float* __restrict__ adj, const float* __restrict__ inBond,
       const float* __restrict__ soft, const unsigned int* __restrict__ WhBf,
       float* __restrict__ mbNext) {
  __shared__ unsigned int lA[64][17];
  __shared__ unsigned int lB[2][64][17];
  const int bsn = blockIdx.x;                 // bs*64 + n
  const int bs = bsn >> 6, n = bsn & 63;
  const int g0 = blockIdx.y * 64;
  const int tid = threadIdx.x, lane = tid & 31, wave = tid >> 5;
  const float* mrow = msgA + (long)bsn * Hd;               // msgA[bs,n,:]
  const float* adjc = adj + bs * Nd * Nd + n;              // adj[bs,m,n] (+m*64)
  const float* mbb  = mbCur + (long)bs * MB_BS_STRIDE + n * Hd; // mb[bs,m,n,:]
  const unsigned int* wbase = WhBf + (long)g0 * PH;        // bf16-pair weights
  v8f acc[4] = {};
  if (wave == 0) tdm_load_w(lds_off(&lB[0][0][0]), wbase); // prologue DMA, kc=0
  for (int kc = 0; kc < 8; ++kc) {            // K = 256
    const int k0 = kc * 32;
    for (int p = tid; p < 1024; p += 128) {   // build transposed A on the fly
      int m = p >> 4, kp = p & 15, k = k0 + (kp << 1);
      float av = adjc[m * Nd];
      const float* mbr = mbb + (long)m * MB_N_STRIDE;
      float x0 = av * mrow[k]     - mbr[k];
      float x1 = av * mrow[k + 1] - mbr[k + 1];
      lA[m][kp] = pack2bf(x0, x1);
      if (kc + 1 < 8) __builtin_prefetch(mbr + k + 32, 0, 1);  // global_prefetch_b8
    }
    if (wave == 0) {
      if (kc < 7) {                           // DMA next slab into other buffer
        tdm_load_w(lds_off(&lB[(kc + 1) & 1][0][0]), wbase + (kc + 1) * 16);
        __builtin_amdgcn_s_wait_tensorcnt(1); // current slab's DMA is done
      } else {
        __builtin_amdgcn_s_wait_tensorcnt(0);
      }
    }
    __syncthreads();
    wmma_4x(&lA[0][0], 17, &lB[kc & 1][0][0], 17, 0, lane, wave * 16, acc);
    __syncthreads();
  }
  const int lr = lane & 15, hi = lane >> 4;
#pragma unroll
  for (int c = 0; c < 4; ++c)
#pragma unroll
    for (int v = 0; v < 8; ++v) {
      int m = wave * 16 + v + (hi << 3);
      int g = g0 + (c << 4) + lr;
      long r = (long)bsn * Nd + m;
      float val = inBond[r * Hd + g] + acc[c][v];
      val = val > 0.f ? val : 0.f;
      mbNext[r * Hd + g] = val * soft[bs * Nd * Nd + n * Nd + m];
    }
}

// ---- out = relu([agg, msgA, inAtom] @ W_o^T + b) ---------------------------
__global__ void __launch_bounds__(128)
k_out(const float* __restrict__ aggB, const float* __restrict__ msgA,
      const float* __restrict__ inAtom, const unsigned int* __restrict__ WoBf,
      const float* __restrict__ bo, float* __restrict__ out) {
  __shared__ unsigned int lA[64][17], lB[64][17];
  const int tid = threadIdx.x, lane = tid & 31, wave = tid >> 5;
  const int m0 = blockIdx.x * 64, n0 = blockIdx.y * 64;
  v8f acc[4] = {};
  for (int kc = 0; kc < 24; ++kc) {           // K = 768 = 3 concatenated 256s
    const int k0 = kc * 32;
    const float* src; int kk;
    if (k0 < 256)      { src = aggB;   kk = k0;       }
    else if (k0 < 512) { src = msgA;   kk = k0 - 256; }
    else               { src = inAtom; kk = k0 - 512; }
    stage64x32(lA, src + (long)m0 * Hd, Hd, kk, Hd, tid);
    stage64x32_bf(lB, WoBf + (long)n0 * PO, PO, kc * 16, PO, tid);
    __syncthreads();
    wmma_4x(&lA[0][0], 17, &lB[0][0], 17, 0, lane, wave * 16, acc);
    __syncthreads();
  }
  const int lr = lane & 15, hi = lane >> 4;
#pragma unroll
  for (int c = 0; c < 4; ++c)
#pragma unroll
    for (int v = 0; v < 8; ++v) {
      int row = m0 + wave * 16 + v + (hi << 3);
      int g   = n0 + (c << 4) + lr;
      float val = acc[c][v] + bo[g];
      out[row * Hd + g] = val > 0.f ? val : 0.f;
    }
}

// ---------------------------------------------------------------------------
extern "C" void kernel_launch(void* const* d_in, const int* in_sizes, int n_in,
                              void* d_out, int out_size, void* d_ws, size_t ws_size,
                              hipStream_t stream) {
  (void)in_sizes; (void)n_in; (void)out_size; (void)ws_size;
  const float* f_atoms  = (const float*)d_in[0];
  const float* f_bonds  = (const float*)d_in[1];
  const float* adj      = (const float*)d_in[2];
  const float* W_i_atom = (const float*)d_in[3];
  const float* W_i_bond = (const float*)d_in[4];
  const float* W_h_0    = (const float*)d_in[5];
  const float* W_h_1    = (const float*)d_in[6];
  const float* W_o_w    = (const float*)d_in[7];
  const float* W_o_b    = (const float*)d_in[8];

  float* ws     = (float*)d_ws;
  float* inAtom = ws;                         // 1024*256
  float* msgA   = inAtom + 262144;            // 1024*256
  float* aggB   = msgA + 262144;              // 1024*256
  float* soft   = aggB + 262144;              // 16*64*64
  float* inBond = soft + 65536;               // 65536*256
  float* mb1    = inBond + 16777216;          // 65536*256
  unsigned int* wAtomBf = (unsigned int*)(mb1 + 16777216);  // 256*67
  unsigned int* wBondBf = wAtomBf + 256 * PA;               // 256*74
  unsigned int* wh0Bf   = wBondBf + 256 * PB;               // 256*128
  unsigned int* wh1Bf   = wh0Bf   + 256 * PH;               // 256*128
  unsigned int* woBf    = wh1Bf   + 256 * PH;               // 256*384
  float* outA   = (float*)d_out;              // atom_hiddens [1024,256]
  float* mbOut  = outA + 262144;              // message_bond [65536,256]

  // pre-pack all weights to bf16 pairs (tiny, L2-resident afterwards)
  k_cvt<<<(256 * PA + 255) / 256, 256, 0, stream>>>(W_i_atom, wAtomBf, AFd, PA, 256 * PA);
  k_cvt<<<(256 * PB + 255) / 256, 256, 0, stream>>>(W_i_bond, wBondBf, BFd, PB, 256 * PB);
  k_cvt<<<(256 * PH + 255) / 256, 256, 0, stream>>>(W_h_0,    wh0Bf,   Hd,  PH, 256 * PH);
  k_cvt<<<(256 * PH + 255) / 256, 256, 0, stream>>>(W_h_1,    wh1Bf,   Hd,  PH, 256 * PH);
  k_cvt<<<(256 * PO + 255) / 256, 256, 0, stream>>>(W_o_w,    woBf,   768,  PO, 256 * PO);

  const dim3 blk(128);
  k_in_atom<<<dim3(16, 4),   blk, 0, stream>>>(f_atoms, wAtomBf, inAtom, msgA);
  k_in_bond<<<dim3(1024, 4), blk, 0, stream>>>(f_bonds, wBondBf, adj, inBond);

  // step 0 (message_bond == input_bond)
  k_agg<<<1024, 256, 0, stream>>>(inBond, msgA, aggB, 0);
  k_resonance<<<16, blk, 0, stream>>>(msgA, adj, soft);
  k_step<<<dim3(1024, 4), blk, 0, stream>>>(inBond, msgA, adj, inBond, soft, wh0Bf, mb1);

  // step 1
  k_agg<<<1024, 256, 0, stream>>>(mb1, msgA, aggB, 0);
  k_resonance<<<16, blk, 0, stream>>>(msgA, adj, soft);
  k_step<<<dim3(1024, 4), blk, 0, stream>>>(mb1, msgA, adj, inBond, soft, wh1Bf, mbOut);

  // final aggregation + output layer
  k_agg<<<1024, 256, 0, stream>>>(mbOut, msgA, aggB, 1);
  k_out<<<dim3(16, 4), blk, 0, stream>>>(aggB, msgA, inAtom, W_o_w ? woBf : woBf, W_o_b, outA);
}
